// BertLayer_28106265985372
// MI455X (gfx1250) — compile-verified, matrix operand order checked by reading, not run on hardware
//
#include <hip/hip_runtime.h>
#include <hip/hip_bf16.h>

// ---------------------------------------------------------------------------
// Types / helpers
// ---------------------------------------------------------------------------
typedef __attribute__((ext_vector_type(16))) __bf16 bf16x16;
typedef __attribute__((ext_vector_type(8)))  float  v8f;
typedef __attribute__((ext_vector_type(4)))  int    v4i;

typedef __attribute__((address_space(1))) v4i* as1_v4i_p;
typedef __attribute__((address_space(3))) v4i* as3_v4i_p;

union Frag {
    bf16x16        v;
    uint4          q[2];
    unsigned short u[16];
};

__device__ __forceinline__ unsigned short f2bf(float f) {
    unsigned u = __builtin_bit_cast(unsigned, f);
    unsigned r = u + 0x7FFFu + ((u >> 16) & 1u);   // round-to-nearest-even
    return (unsigned short)(r >> 16);
}

__device__ __forceinline__ v8f wmma_bf16(const Frag& a, const Frag& b, v8f c) {
    return __builtin_amdgcn_wmma_f32_16x16x32_bf16(
        /*neg_a=*/false, a.v, /*neg_b=*/false, b.v,
        /*c_mod=*/(short)0, c, /*reuse_a=*/false, /*reuse_b=*/false);
}

// reductions across the 16 lanes of a half-wave (wave32; masks 1..8 stay in half)
__device__ __forceinline__ float halfmax(float v) {
    #pragma unroll
    for (int m = 1; m < 16; m <<= 1) v = fmaxf(v, __shfl_xor(v, m, 32));
    return v;
}
__device__ __forceinline__ float halfsum(float v) {
    #pragma unroll
    for (int m = 1; m < 16; m <<= 1) v += __shfl_xor(v, m, 32);
    return v;
}

// ---------------------------------------------------------------------------
// Async global -> LDS copy (16B per lane), with synchronous fallback.
// gfx1250: GLOBAL_LOAD_ASYNC_TO_LDS_B128, tracked by ASYNCcnt.
// ---------------------------------------------------------------------------
#if defined(__gfx1250__) && __has_builtin(__builtin_amdgcn_global_load_async_to_lds_b128)
#define HAS_ASYNC_LDS 1
#endif

__device__ __forceinline__ void copy16(unsigned short* lds, const unsigned short* glb) {
#ifdef HAS_ASYNC_LDS
    __builtin_amdgcn_global_load_async_to_lds_b128(
        (as1_v4i_p)(glb), (as3_v4i_p)(lds), /*offset=*/0, /*cpol=*/0);
#else
    *(uint4*)lds = *(const uint4*)glb;
#endif
}

__device__ __forceinline__ void async_wait() {
#ifdef HAS_ASYNC_LDS
#if __has_builtin(__builtin_amdgcn_s_wait_asynccnt)
    __builtin_amdgcn_s_wait_asynccnt(0);
#else
    asm volatile("s_wait_asynccnt 0" ::: "memory");
#endif
#endif
}

// ---------------------------------------------------------------------------
// f32 -> bf16 conversion
// ---------------------------------------------------------------------------
__global__ void cvt_f32_bf16(const float* __restrict__ in,
                             unsigned short* __restrict__ out, size_t n) {
    size_t i = (size_t)blockIdx.x * blockDim.x + threadIdx.x;
    if (i < n) out[i] = f2bf(in[i]);
}

// ---------------------------------------------------------------------------
// Tiled WMMA GEMM:  C[M,N] = A[M,K] * B[N,K]^T (+bias) (+GELU)
// A, B bf16 row-major (K contiguous).  WG: 256 thr = 8 waves, tile 64(M)x128(N),
// K-step 32, double-buffered LDS with async global->LDS staging pipeline.
// Each wave -> 16x64 output via 4 f32 accumulators.
// ---------------------------------------------------------------------------
template<bool OUT_BF16, bool ACT_GELU>
__global__ __launch_bounds__(256) void gemm_bf16_kernel(
    const unsigned short* __restrict__ A, const unsigned short* __restrict__ B,
    const float* __restrict__ bias,
    float* __restrict__ Cf, unsigned short* __restrict__ Cb,
    int M, int N, int K)
{
    __shared__ unsigned short As[2][64][32];
    __shared__ unsigned short Bs[2][128][32];

    const int tid  = threadIdx.x;
    const int lane = tid & 31;
    const int wave = tid >> 5;
    const int wm   = wave & 3;      // M sub-block (4)
    const int wn   = wave >> 2;     // N sub-block (2)
    const int half = (lane >> 4) & 1;
    const int l15  = lane & 15;
    const size_t mBase = (size_t)blockIdx.y * 64;
    const size_t nBase = (size_t)blockIdx.x * 128;

    // per-thread staging slots
    const int ra = tid >> 2, ca = (tid & 3) * 8;    // A: 64 rows x 4 chunks of 8
    const int rb = tid >> 1, cb = (tid & 1) * 16;   // B: 128 rows x 2 chunks of 16
    const unsigned short* Aptr = A + (mBase + ra) * (size_t)K + ca;
    const unsigned short* Bptr = B + (nBase + rb) * (size_t)K + cb;

    auto stage = [&](int buf, int kt) {
        copy16(&As[buf][ra][ca],     Aptr + kt);
        copy16(&Bs[buf][rb][cb],     Bptr + kt);
        copy16(&Bs[buf][rb][cb + 8], Bptr + kt + 8);
    };

    v8f acc[4] = {};

    stage(0, 0);
    int buf = 0;
    for (int kt = 0; kt < K; kt += 32, buf ^= 1) {
        async_wait();
        __syncthreads();
        if (kt + 32 < K) stage(buf ^ 1, kt + 32);
        if (kt + 64 < K) {   // pull line for the tile after next into GL2
            __builtin_prefetch(Aptr + kt + 64, 0, 0);
            __builtin_prefetch(Bptr + kt + 64, 0, 0);
        }

        // A fragment: 16x32, lane = M row; K split per half-wave
        Frag a;
        {
            const unsigned short* rp = &As[buf][wm * 16 + l15][0];
            a.q[0] = *(const uint4*)(rp + half * 8);
            a.q[1] = *(const uint4*)(rp + half * 8 + 16);
        }
        #pragma unroll
        for (int nt = 0; nt < 4; ++nt) {
            Frag bfr;
            const unsigned short* rp = &Bs[buf][wn * 64 + nt * 16 + l15][0];
            bfr.q[0] = *(const uint4*)(rp + half * 8);
            bfr.q[1] = *(const uint4*)(rp + half * 8 + 16);
            acc[nt] = wmma_bf16(a, bfr, acc[nt]);
        }
    }

    // epilogue: C element (M = r + 8*half, N = l15) per VGPR r
    #pragma unroll
    for (int nt = 0; nt < 4; ++nt) {
        size_t col = nBase + wn * 64 + nt * 16 + l15;
        float bv = bias ? bias[col] : 0.0f;
        #pragma unroll
        for (int r = 0; r < 8; ++r) {
            size_t row = mBase + wm * 16 + r + 8 * half;
            float x = acc[nt][r] + bv;
            if (ACT_GELU) x = 0.5f * x * (1.0f + erff(x * 0.70710678118f));
            if (OUT_BF16) Cb[row * (size_t)N + col] = f2bf(x);
            else          Cf[row * (size_t)N + col] = x;
        }
    }
}

// ---------------------------------------------------------------------------
// Flash attention.  Grid: (S/64, H, B), 128 threads = 4 waves.
// Wave handles 16 query rows x DK=64; streams 32-key blocks.
// K/V blocks double-buffered in LDS via async copies.
// q,k,v,out are bf16 [B*S, H*64].
// ---------------------------------------------------------------------------
__global__ __launch_bounds__(128) void attn_kernel(
    const unsigned short* __restrict__ Q, const unsigned short* __restrict__ Kg,
    const unsigned short* __restrict__ Vg, const unsigned char* __restrict__ mask,
    unsigned short* __restrict__ O, int S, int Hn)
{
    __shared__ unsigned short Ks[2][32][64];
    __shared__ unsigned short Vs[2][32][64];
    __shared__ unsigned short Ps[4][16][32];

    const int tid   = threadIdx.x;
    const int lane  = tid & 31;
    const int wave  = tid >> 5;
    const int half  = (lane >> 4) & 1;
    const int l15   = lane & 15;
    const int b     = blockIdx.z;
    const int h     = blockIdx.y;
    const int qbase = blockIdx.x * 64 + wave * 16;
    const size_t rs = (size_t)Hn * 64;   // 512

    // staging slots: 32 rows x 4 chunks of 16
    const int rr = tid >> 2, rc = (tid & 3) * 16;
    auto stageKV = [&](int buf, int j) {
        size_t src = (size_t)(b * S + j + rr) * rs + h * 64 + rc;
        copy16(&Ks[buf][rr][rc],     Kg + src);
        copy16(&Ks[buf][rr][rc + 8], Kg + src + 8);
        copy16(&Vs[buf][rr][rc],     Vg + src);
        copy16(&Vs[buf][rr][rc + 8], Vg + src + 8);
    };

    // Q fragments (16 rows x 64, two K=32 chunks)
    Frag aq[2];
    {
        const unsigned short* rp =
            Q + ((size_t)(b * S + qbase + l15) * rs + h * 64);
        #pragma unroll
        for (int c = 0; c < 2; ++c) {
            aq[c].q[0] = *(const uint4*)(rp + c * 32 + half * 8);
            aq[c].q[1] = *(const uint4*)(rp + c * 32 + half * 8 + 16);
        }
    }

    float mrun[8], lrun[8];
    v8f accO[4] = {};
    #pragma unroll
    for (int r = 0; r < 8; ++r) { mrun[r] = -1e30f; lrun[r] = 0.0f; }

    stageKV(0, 0);
    int buf = 0;
    for (int j = 0; j < S; j += 32, buf ^= 1) {
        async_wait();
        __syncthreads();
        if (j + 32 < S) stageKV(buf ^ 1, j + 32);

        // scores: two 16-key tiles, K(=DK)=64 via 2 chained WMMAs each
        float st[2][8];
        #pragma unroll
        for (int t = 0; t < 2; ++t) {
            v8f sc = {};
            #pragma unroll
            for (int c = 0; c < 2; ++c) {
                Frag kb;
                const unsigned short* rp = &Ks[buf][t * 16 + l15][c * 32];
                kb.q[0] = *(const uint4*)(rp + half * 8);
                kb.q[1] = *(const uint4*)(rp + half * 8 + 16);
                sc = wmma_bf16(aq[c], kb, sc);
            }
            int key = j + t * 16 + l15;
            float msub = mask[b * S + key] ? 0.0f : 100.0f;
            #pragma unroll
            for (int r = 0; r < 8; ++r) st[t][r] = sc[r] * 0.125f - msub;
        }

        // online softmax (row = r + 8*half; reduce across 16-lane half)
        float pv[2][8];
        #pragma unroll
        for (int r = 0; r < 8; ++r) {
            float rowmax = halfmax(fmaxf(st[0][r], st[1][r]));
            float mnew   = fmaxf(mrun[r], rowmax);
            float cf     = __expf(mrun[r] - mnew);
            mrun[r] = mnew;
            float p0 = __expf(st[0][r] - mnew);
            float p1 = __expf(st[1][r] - mnew);
            pv[0][r] = p0; pv[1][r] = p1;
            lrun[r] = lrun[r] * cf + halfsum(p0 + p1);
            #pragma unroll
            for (int nt = 0; nt < 4; ++nt) accO[nt][r] *= cf;
        }

        // C-layout -> A-layout for P via LDS
        #pragma unroll
        for (int t = 0; t < 2; ++t)
            #pragma unroll
            for (int r = 0; r < 8; ++r)
                Ps[wave][r + 8 * half][t * 16 + l15] = f2bf(pv[t][r]);
        __syncthreads();

        Frag pf;
        {
            const unsigned short* rp = &Ps[wave][l15][0];
            pf.q[0] = *(const uint4*)(rp + half * 8);
            pf.q[1] = *(const uint4*)(rp + half * 8 + 16);
        }
        // P(16x32) x V(32x64): 4 WMMAs
        #pragma unroll
        for (int nt = 0; nt < 4; ++nt) {
            Frag vf;
            #pragma unroll
            for (int i = 0; i < 8; ++i) {
                vf.u[i]     = Vs[buf][half * 8 + i][nt * 16 + l15];
                vf.u[i + 8] = Vs[buf][half * 8 + 16 + i][nt * 16 + l15];
            }
            accO[nt] = wmma_bf16(pf, vf, accO[nt]);
        }
    }

    // normalize and write ctx (bf16)
    #pragma unroll
    for (int nt = 0; nt < 4; ++nt) {
        #pragma unroll
        for (int r = 0; r < 8; ++r) {
            int row = qbase + r + 8 * half;
            float o = accO[nt][r] / lrun[r];
            O[(size_t)(b * S + row) * rs + h * 64 + nt * 16 + l15] = f2bf(o);
        }
    }
}

// ---------------------------------------------------------------------------
// out = LayerNorm(a + res) ; optionally also emit bf16 copy.
// One block per row, 256 threads, D=512 (2 elems/thread).
// ---------------------------------------------------------------------------
template<bool WRITE_BF16>
__global__ __launch_bounds__(256) void add_ln_kernel(
    const float* __restrict__ a, const float* __restrict__ res,
    const float* __restrict__ g, const float* __restrict__ be,
    float* __restrict__ outf, unsigned short* __restrict__ outb, int D)
{
    __shared__ float red[256];
    const size_t row = blockIdx.x;
    const int tid = threadIdx.x;
    const size_t base = row * (size_t)D;

    float v0 = a[base + tid]       + res[base + tid];
    float v1 = a[base + tid + 256] + res[base + tid + 256];

    red[tid] = v0 + v1;
    __syncthreads();
    for (int s = 128; s > 0; s >>= 1) {
        if (tid < s) red[tid] += red[tid + s];
        __syncthreads();
    }
    float mu = red[0] / (float)D;
    __syncthreads();

    red[tid] = (v0 - mu) * (v0 - mu) + (v1 - mu) * (v1 - mu);
    __syncthreads();
    for (int s = 128; s > 0; s >>= 1) {
        if (tid < s) red[tid] += red[tid + s];
        __syncthreads();
    }
    float rstd = rsqrtf(red[0] / (float)D + 1e-5f);

    float o0 = (v0 - mu) * rstd * g[tid]       + be[tid];
    float o1 = (v1 - mu) * rstd * g[tid + 256] + be[tid + 256];
    outf[base + tid]       = o0;
    outf[base + tid + 256] = o1;
    if (WRITE_BF16) {
        outb[base + tid]       = f2bf(o0);
        outb[base + tid + 256] = f2bf(o1);
    }
}

// ---------------------------------------------------------------------------
// Launch
// ---------------------------------------------------------------------------
extern "C" void kernel_launch(void* const* d_in, const int* in_sizes, int n_in,
                              void* d_out, int out_size, void* d_ws, size_t ws_size,
                              hipStream_t stream) {
    (void)in_sizes; (void)n_in; (void)out_size; (void)ws_size;

    const float*         x     = (const float*)d_in[0];
    const unsigned char* mask  = (const unsigned char*)d_in[1];
    const float*         Wq    = (const float*)d_in[2];
    const float*         Wk    = (const float*)d_in[3];
    const float*         Wv    = (const float*)d_in[4];
    const float*         Wp    = (const float*)d_in[5];
    const float*         W1    = (const float*)d_in[6];
    const float*         b1    = (const float*)d_in[7];
    const float*         W2    = (const float*)d_in[8];
    const float*         b2    = (const float*)d_in[9];
    const float*         ln1g  = (const float*)d_in[10];
    const float*         ln1b  = (const float*)d_in[11];
    const float*         ln2g  = (const float*)d_in[12];
    const float*         ln2b  = (const float*)d_in[13];
    float*               out   = (float*)d_out;

    const int Bn = 4, S = 2048, D = 512, Hn = 8, DFF = 2048;
    const size_t MS = (size_t)Bn * S;   // 8192 rows

    // workspace carve-up (256B aligned)
    char* ws = (char*)d_ws;
    size_t off = 0;
    auto alloc = [&](size_t bytes) -> char* {
        char* p = ws + off;
        off = (off + bytes + 255) & ~(size_t)255;
        return p;
    };
    unsigned short* xb   = (unsigned short*)alloc(MS * D * 2);
    unsigned short* wqb  = (unsigned short*)alloc((size_t)D * D * 2);
    unsigned short* wkb  = (unsigned short*)alloc((size_t)D * D * 2);
    unsigned short* wvb  = (unsigned short*)alloc((size_t)D * D * 2);
    unsigned short* wpb  = (unsigned short*)alloc((size_t)D * D * 2);
    unsigned short* w1b  = (unsigned short*)alloc((size_t)DFF * D * 2);
    unsigned short* w2b  = (unsigned short*)alloc((size_t)D * DFF * 2);
    unsigned short* qb   = (unsigned short*)alloc(MS * D * 2);
    unsigned short* kb   = (unsigned short*)alloc(MS * D * 2);
    unsigned short* vb   = (unsigned short*)alloc(MS * D * 2);
    unsigned short* ctxb = (unsigned short*)alloc(MS * D * 2);
    float*          att  = (float*)alloc(MS * D * 4);
    float*          hf   = (float*)alloc(MS * D * 4);
    unsigned short* hb   = (unsigned short*)alloc(MS * D * 2);
    unsigned short* ff1  = (unsigned short*)alloc(MS * DFF * 2);
    float*          ff2  = (float*)alloc(MS * D * 4);

    auto cvt = [&](const float* src, unsigned short* dst, size_t n) {
        cvt_f32_bf16<<<dim3((unsigned)((n + 255) / 256)), dim3(256), 0, stream>>>(src, dst, n);
    };
    cvt(x,  xb,  MS * D);
    cvt(Wq, wqb, (size_t)D * D);
    cvt(Wk, wkb, (size_t)D * D);
    cvt(Wv, wvb, (size_t)D * D);
    cvt(Wp, wpb, (size_t)D * D);
    cvt(W1, w1b, (size_t)DFF * D);
    cvt(W2, w2b, (size_t)D * DFF);

    // Q/K/V projections (bf16 out)
    gemm_bf16_kernel<true, false><<<dim3(D / 128, (unsigned)(MS / 64)), 256, 0, stream>>>(
        xb, wqb, nullptr, nullptr, qb, (int)MS, D, D);
    gemm_bf16_kernel<true, false><<<dim3(D / 128, (unsigned)(MS / 64)), 256, 0, stream>>>(
        xb, wkb, nullptr, nullptr, kb, (int)MS, D, D);
    gemm_bf16_kernel<true, false><<<dim3(D / 128, (unsigned)(MS / 64)), 256, 0, stream>>>(
        xb, wvb, nullptr, nullptr, vb, (int)MS, D, D);

    // attention
    attn_kernel<<<dim3(S / 64, Hn, Bn), 128, 0, stream>>>(qb, kb, vb, mask, ctxb, S, Hn);

    // output projection (f32 out)
    gemm_bf16_kernel<false, false><<<dim3(D / 128, (unsigned)(MS / 64)), 256, 0, stream>>>(
        ctxb, wpb, nullptr, att, nullptr, (int)MS, D, D);

    // h = LN(att + x), also bf16 copy for FFN
    add_ln_kernel<true><<<dim3((unsigned)MS), 256, 0, stream>>>(att, x, ln1g, ln1b, hf, hb, D);

    // FFN
    gemm_bf16_kernel<true, true><<<dim3(DFF / 128, (unsigned)(MS / 64)), 256, 0, stream>>>(
        hb, w1b, b1, nullptr, ff1, (int)MS, DFF, D);
    gemm_bf16_kernel<false, false><<<dim3(D / 128, (unsigned)(MS / 64)), 256, 0, stream>>>(
        ff1, w2b, b2, ff2, nullptr, (int)MS, D, DFF);

    // out = LN(ff2 + h)
    add_ln_kernel<false><<<dim3((unsigned)MS), 256, 0, stream>>>(ff2, hf, ln2g, ln2b, out, nullptr, D);
}
